// CategorySpecificLinear_19954418057650
// MI455X (gfx1250) — compile-verified, matrix operand order checked by reading, not run on hardware
//
#include <hip/hip_runtime.h>

typedef __attribute__((ext_vector_type(16))) __bf16        v16bf;
typedef __attribute__((ext_vector_type(8)))  float         v8f;
typedef __attribute__((ext_vector_type(8)))  unsigned int  v8u;

#define TM 128
#define TN 128
#define TK 32
#define APITCH 17      // 16 u32 (32 bf16) per row + 1 pad
#define NTHREADS 256   // 8 waves (wave32): 4(M) x 2(N)

__device__ __forceinline__ unsigned pack_bf16(float a, float b) {
    // round-to-nearest-even fp32 -> bf16, packed pair (low = first arg)
    unsigned ua = __float_as_uint(a);
    unsigned ub = __float_as_uint(b);
    ua += 0x7FFFu + ((ua >> 16) & 1u);
    ub += 0x7FFFu + ((ub >> 16) & 1u);
    return (ua >> 16) | (ub & 0xFFFF0000u);
}

// ---------------- Phase 1: counting sort by category ----------------

__global__ void csl_zero(int* counts, int* cursor, int C) {
    int i = threadIdx.x;
    if (i < C) { counts[i] = 0; cursor[i] = 0; }
}

__global__ void csl_hist(const int* __restrict__ cid, int* counts, int T) {
    int t = blockIdx.x * blockDim.x + threadIdx.x;
    if (t < T) atomicAdd(&counts[cid[t]], 1);
}

__global__ void csl_scan(const int* counts, int* offsets, int C) {
    if (threadIdx.x == 0) {
        int s = 0;
        for (int c = 0; c < C; ++c) { offsets[c] = s; s += counts[c]; }
        offsets[C] = s;
    }
}

__global__ void csl_scatter(const int* __restrict__ cid, const int* __restrict__ offsets,
                            int* cursor, int* perm, int T) {
    int t = blockIdx.x * blockDim.x + threadIdx.x;
    if (t < T) {
        int c = cid[t];
        int pos = offsets[c] + atomicAdd(&cursor[c], 1);
        perm[pos] = t;
    }
}

// ---------------- Phase 2: grouped GEMM with bf16 WMMA ----------------
// block tile: 128 tokens x 128 outputs; 8 waves as 4(M) x 2(N),
// each wave 32(M) x 64(N): 2 A-frags x 4 B-frags -> 8 WMMAs/K-step.

__global__ __launch_bounds__(NTHREADS)
void csl_gemm(const float* __restrict__ x, const float* __restrict__ weight,
              const float* __restrict__ bias, const int* __restrict__ offsets,
              const int* __restrict__ perm, float* __restrict__ out,
              int D, int O) {
    const int c   = blockIdx.z;
    const int off = offsets[c];
    const int cnt = offsets[c + 1] - off;
    const int m0  = blockIdx.y * TM;
    if (m0 >= cnt) return;
    const int n0  = blockIdx.x * TN;

    __shared__ unsigned As[TM][APITCH];  // A: [row M][K-pair] packed bf16
    __shared__ unsigned Bs[TN][APITCH];  // B: [col N][K-pair] packed bf16

    const int tid = threadIdx.x;

    // A-tile loader coords: thread -> (row, K-half): 128 rows x 2 halves
    const int ar   = tid >> 1;   // 0..127 token row within tile
    const int ah   = tid & 1;    // 0/1 -> K offset 0/16
    const int arow = m0 + ar;
    const float* aptr = nullptr;
    if (arow < cnt) {
        int tok = perm[off + arow];
        aptr = x + (size_t)tok * D + ah * 16;
    }

    // B-tile loader coords: thread -> (K pair, 8 N cols): 16 x 16 chunks
    const int bk = (tid >> 4) << 1;  // 0,2,..,30
    const int bn = (tid & 15) << 3;  // 0,8,..,120
    const float* bptr = weight + ((size_t)c * D + bk) * O + (n0 + bn);

    const int lane = tid & 31;
    const int l  = lane & 15;
    const int h  = lane >> 4;
    const int w  = tid >> 5;         // wave id 0..7
    const int wm = w >> 1;           // 0..3 -> M base wm*32
    const int wn = w & 1;            // 0..1 -> N base wn*64

    v8f acc[2][4] = {};

    for (int k0 = 0; k0 < D; k0 += TK) {
        // ---- stage A tile (fp32 -> bf16 pairs) ----
        {
            float4 f0, f1, f2, f3;
            if (aptr) {
                const float4* p = (const float4*)(aptr + k0);
                f0 = p[0]; f1 = p[1]; f2 = p[2]; f3 = p[3];
            } else {
                f0 = f1 = f2 = f3 = make_float4(0.f, 0.f, 0.f, 0.f);
            }
            unsigned* dst = &As[ar][ah * 8];
            dst[0] = pack_bf16(f0.x, f0.y);
            dst[1] = pack_bf16(f0.z, f0.w);
            dst[2] = pack_bf16(f1.x, f1.y);
            dst[3] = pack_bf16(f1.z, f1.w);
            dst[4] = pack_bf16(f2.x, f2.y);
            dst[5] = pack_bf16(f2.z, f2.w);
            dst[6] = pack_bf16(f3.x, f3.y);
            dst[7] = pack_bf16(f3.z, f3.w);
        }
        // ---- stage B tile, transposed to [N][K-pair] ----
        {
            const float* r0 = bptr + (size_t)k0 * O;
            const float* r1 = r0 + O;
            float4 a0 = ((const float4*)r0)[0];
            float4 a1 = ((const float4*)r0)[1];
            float4 b0 = ((const float4*)r1)[0];
            float4 b1 = ((const float4*)r1)[1];
            const int kc = bk >> 1;
            Bs[bn + 0][kc] = pack_bf16(a0.x, b0.x);
            Bs[bn + 1][kc] = pack_bf16(a0.y, b0.y);
            Bs[bn + 2][kc] = pack_bf16(a0.z, b0.z);
            Bs[bn + 3][kc] = pack_bf16(a0.w, b0.w);
            Bs[bn + 4][kc] = pack_bf16(a1.x, b1.x);
            Bs[bn + 5][kc] = pack_bf16(a1.y, b1.y);
            Bs[bn + 6][kc] = pack_bf16(a1.z, b1.z);
            Bs[bn + 7][kc] = pack_bf16(a1.w, b1.w);
        }
        __syncthreads();

        // ---- fragments per ISA layout: lane l = M (A) / N (B), h picks K-halves ----
        v16bf a[2];
        #pragma unroll
        for (int s = 0; s < 2; ++s) {
            v8u au;
            const unsigned* ap = &As[wm * 32 + s * 16 + l][0];
            #pragma unroll
            for (int j = 0; j < 4; ++j) au[j]     = ap[h * 4 + j];       // K = h*8 + 0..7
            #pragma unroll
            for (int j = 0; j < 4; ++j) au[4 + j] = ap[8 + h * 4 + j];   // K = 16 + h*8 + 0..7
            a[s] = __builtin_bit_cast(v16bf, au);
        }

        #pragma unroll
        for (int f = 0; f < 4; ++f) {
            v8u bu;
            const unsigned* bp = &Bs[wn * 64 + f * 16 + l][0];
            #pragma unroll
            for (int j = 0; j < 4; ++j) bu[j]     = bp[h * 4 + j];
            #pragma unroll
            for (int j = 0; j < 4; ++j) bu[4 + j] = bp[8 + h * 4 + j];
            v16bf b = __builtin_bit_cast(v16bf, bu);
            #pragma unroll
            for (int s = 0; s < 2; ++s) {
                acc[s][f] = __builtin_amdgcn_wmma_f32_16x16x32_bf16(
                    false, a[s], false, b, (short)0, acc[s][f], false, false);
            }
        }
        __syncthreads();
    }

    // ---- epilogue: C/D layout acc[v] = C[M = v + 8h][N = l]; scatter + bias ----
    #pragma unroll
    for (int f = 0; f < 4; ++f) {
        const int col = n0 + wn * 64 + f * 16 + l;
        const float bv = bias[(size_t)c * O + col];
        #pragma unroll
        for (int s = 0; s < 2; ++s) {
            #pragma unroll
            for (int v = 0; v < 8; ++v) {
                const int row = m0 + wm * 32 + s * 16 + h * 8 + v;
                if (row < cnt) {
                    const int tok = perm[off + row];
                    out[(size_t)tok * O + col] = acc[s][f][v] + bv;
                }
            }
        }
    }
}

// ---------------- host launcher ----------------

extern "C" void kernel_launch(void* const* d_in, const int* in_sizes, int n_in,
                              void* d_out, int out_size, void* d_ws, size_t ws_size,
                              hipStream_t stream) {
    const float* x      = (const float*)d_in[0];
    const int*   cid    = (const int*)d_in[1];
    const float* weight = (const float*)d_in[2];
    const float* bias   = (const float*)d_in[3];
    float*       out    = (float*)d_out;

    const int T = in_sizes[1];               // B*S tokens
    const int D = in_sizes[0] / T;           // 1024
    const int O = out_size / T;              // 1024
    const int C = in_sizes[3] / O;           // 8

    int* ws      = (int*)d_ws;
    int* counts  = ws;                       // C
    int* offsets = ws + C;                   // C+1
    int* cursor  = ws + 2 * C + 1;           // C
    int* perm    = ws + 3 * C + 2;           // T

    csl_zero<<<1, 64, 0, stream>>>(counts, cursor, C);
    csl_hist<<<(T + 255) / 256, 256, 0, stream>>>(cid, counts, T);
    csl_scan<<<1, 32, 0, stream>>>(counts, offsets, C);
    csl_scatter<<<(T + 255) / 256, 256, 0, stream>>>(cid, offsets, cursor, perm, T);

    dim3 grid(O / TN, (T + TM - 1) / TM, C);
    csl_gemm<<<grid, NTHREADS, 0, stream>>>(x, weight, bias, offsets, perm, out, D, O);
}
